// P4Net_36618891165873
// MI455X (gfx1250) — matrix operand order repaired and checked by reading, not compiled, not run on hardware
//
#include <hip/hip_runtime.h>

// SNN LIF forward, 5 layers, T=20, WMMA f16->f32 on gfx1250 (wave32).
// State variable is the pre-leaked accumulator cpre = 0.9*mem - spike(mem):
//   d = wmma(A, B, cpre); fire = d > 1; cpre' = 0.9*d - fire.
// Per neuron per step: 1 v_cmp + 1 v_cndmask_b16 + 1 v_fma_mix.
// All 12 weight B-fragments are hoisted into registers for the whole T loop;
// x loads are software-pipelined one step ahead. No workgroup barriers:
// activation staging is wave-private LDS (in-order within a wave).

typedef __attribute__((ext_vector_type(16))) _Float16 v16h;
typedef __attribute__((ext_vector_type(8)))  _Float16 v8h;
typedef __attribute__((ext_vector_type(8)))  float    v8f;

#define TSTEPS 20
#define BETA   0.9f
#define THRESH 1.0f

// ---- LDS layout (halves) ----
#define OFF_W1 0        // [64][32]
#define OFF_W2 2048     // [32][64]
#define OFF_W3 4096     // [32][32]
#define OFF_W4 5120     // [16][32]
#define OFF_W5 5632     // [16][32] rows 6..15 zero, cols 16..31 zero
#define OFF_ST 6144     // 8 waves * 16 rows * 64 cols
#define SMEM_HALVES (OFF_ST + 8*16*64)

__device__ __forceinline__ v16h cat16(v8h lo, v8h hi) {
  return __builtin_shufflevector(lo, hi, 0,1,2,3,4,5,6,7,8,9,10,11,12,13,14,15);
}

// A fragment: 16(M) x 32(K) f16, ISA layout.
// lane<16 : row M=lane,    K = {0..7, 16..23}
// lane>=16: row M=lane-16, K = {8..15, 24..31}
__device__ __forceinline__ v16h load_a_frag(const _Float16* st, int lane, int koff) {
  const int m  = lane & 15;
  const int kh = (lane & 16) ? 8 : 0;
  v8h lo = *(const v8h*)(st + m*64 + koff + kh);
  v8h hi = *(const v8h*)(st + m*64 + koff + kh + 16);
  return cat16(lo, hi);
}

// B fragment: 32(K) x 16(N) f16. W stored [N][Kpad]:
// lane<16 -> N=lane, K=koff+{0..7,16..23} ; lane>=16 -> N=lane-16, K=koff+{8..15,24..31}
__device__ __forceinline__ v16h load_b_frag(const _Float16* w, int lane, int kpad,
                                            int ntile, int koff) {
  const int n  = (lane & 15) + (ntile << 4);
  const int kk = koff + ((lane & 16) ? 8 : 0);
  v8h lo = *(const v8h*)(w + n*kpad + kk);
  v8h hi = *(const v8h*)(w + n*kpad + kk + 16);
  return cat16(lo, hi);
}

// Post-WMMA LIF: write f16 spikes to the stage tile, return cpre' = 0.9*d - spike.
__device__ __forceinline__ v8f fire_leak(v8f d, _Float16* st, int lane, int ntile) {
  const int n  = (lane & 15) + (ntile << 4);
  const int mb = (lane & 16) ? 8 : 0;
  v8f cn;
#pragma unroll
  for (int r = 0; r < 8; ++r) {
    const float m = d[r];
    const _Float16 h = (m > THRESH) ? (_Float16)1.0f : (_Float16)0.0f;
    st[(mb + r)*64 + n] = h;
    cn[r] = BETA * m - (float)h;     // -> v_fma_mix_f32 (f16 operand folded)
  }
  return cn;
}

__device__ __forceinline__ void fill_w(_Float16* dst, const float* src,
                                       int N, int K, int kpad, int tid) {
  for (int idx = tid; idx < N*K; idx += 256)
    dst[(idx / K) * kpad + (idx % K)] = (_Float16)src[idx];
}

__global__ __launch_bounds__(256, 1)
void snn_lif_wmma(const float* __restrict__ xg,
                  const float* __restrict__ w1g, const float* __restrict__ w2g,
                  const float* __restrict__ w3g, const float* __restrict__ w4g,
                  const float* __restrict__ w5g,
                  float* __restrict__ outg, int Btot)
{
  __shared__ __align__(16) _Float16 smem[SMEM_HALVES];
  __shared__ __align__(16) float    ostage[8][96];   // per-wave 16 samples x 6

  const int tid  = threadIdx.x;
  const int lane = tid & 31;
  const int wave = tid >> 5;

  // ---- init LDS: zero everything (covers all padding), fill weights ----
  {
    unsigned* z = (unsigned*)smem;
    for (int i = tid; i < SMEM_HALVES / 2; i += 256) z[i] = 0u;
  }
  __syncthreads();
  fill_w(smem + OFF_W1, w1g, 64, 10, 32, tid);
  fill_w(smem + OFF_W2, w2g, 32, 64, 64, tid);
  fill_w(smem + OFF_W3, w3g, 32, 32, 32, tid);
  fill_w(smem + OFF_W4, w4g, 16, 32, 32, tid);
  fill_w(smem + OFF_W5, w5g,  6, 16, 32, tid);
  __syncthreads();

  _Float16* st = smem + OFF_ST + wave * (16 * 64);
  float*    ow = &ostage[wave][0];
  const int b0w = blockIdx.x * 128 + wave * 16;   // first sample of this wave
  const size_t xstep = (size_t)Btot * 10;

  // ---- hoist all weight B-fragments into registers for the whole T loop ----
  v16h b1f[4], b2f[2][2], b3f[2], b4f, b5f;
#pragma unroll
  for (int nt = 0; nt < 4; ++nt) b1f[nt] = load_b_frag(smem + OFF_W1, lane, 32, nt, 0);
#pragma unroll
  for (int nt = 0; nt < 2; ++nt) {
    b2f[nt][0] = load_b_frag(smem + OFF_W2, lane, 64, nt, 0);
    b2f[nt][1] = load_b_frag(smem + OFF_W2, lane, 64, nt, 32);
    b3f[nt]    = load_b_frag(smem + OFF_W3, lane, 32, nt, 0);
  }
  b4f = load_b_frag(smem + OFF_W4, lane, 32, 0, 0);
  b5f = load_b_frag(smem + OFF_W5, lane, 32, 0, 0);

  // Pre-leaked accumulator state (cpre), resident in registers for all T.
  v8f cp1[4] = {};
  v8f cp2[2] = {};
  v8f cp3[2] = {};
  v8f cp4    = {};
  v8f cp5    = {};

  // ---- software-pipelined x: preload t=0 into registers ----
  float xr[5];
  {
    const size_t xbase = (size_t)b0w * 10;
#pragma unroll
    for (int j = 0; j < 5; ++j) xr[j] = xg[xbase + j * 32 + lane];
  }

  for (int t = 0; t < TSTEPS; ++t) {
    // ---- stage x_t from registers (16 samples x 10 features, f32->f16) ----
#pragma unroll
    for (int j = 0; j < 5; ++j) {
      const int idx = j * 32 + lane;                 // 0..159
      st[(idx / 10) * 64 + (idx % 10)] = (_Float16)xr[j];
    }

    // Issue next timestep's x loads now; consumed at the top of t+1.
    if (t + 1 < TSTEPS) {
      const size_t xn = ((size_t)(t + 1) * Btot + b0w) * 10;
#pragma unroll
      for (int j = 0; j < 5; ++j) xr[j] = xg[xn + j * 32 + lane];
      if ((t + 2 < TSTEPS) && (lane < 5))
        __builtin_prefetch(xg + xn + xstep + lane * 32, 0, 0);
    }

    // ---- layer 1: K=10(pad32), N=64 (4 N-tiles) ----
    {
      v16h a = load_a_frag(st, lane, 0);
#pragma unroll
      for (int nt = 0; nt < 4; ++nt)
        cp1[nt] = __builtin_amdgcn_wmma_f32_16x16x32_f16(false, a, false, b1f[nt],
                                                         (short)0, cp1[nt], false, false);
#pragma unroll
      for (int nt = 0; nt < 4; ++nt) cp1[nt] = fire_leak(cp1[nt], st, lane, nt);
    }

    // ---- layer 2: K=64 (2 K-tiles), N=32 (2 N-tiles) ----
    {
      v16h a0 = load_a_frag(st, lane, 0);
      v16h a1 = load_a_frag(st, lane, 32);
#pragma unroll
      for (int nt = 0; nt < 2; ++nt) {
        v8f c = __builtin_amdgcn_wmma_f32_16x16x32_f16(false, a0, false, b2f[nt][0],
                                                       (short)0, cp2[nt], false, false);
        cp2[nt] = __builtin_amdgcn_wmma_f32_16x16x32_f16(false, a1, false, b2f[nt][1],
                                                         (short)0, c, false, false);
      }
#pragma unroll
      for (int nt = 0; nt < 2; ++nt) cp2[nt] = fire_leak(cp2[nt], st, lane, nt);
    }

    // ---- layer 3: K=32, N=32 (2 N-tiles) ----
    {
      v16h a = load_a_frag(st, lane, 0);
#pragma unroll
      for (int nt = 0; nt < 2; ++nt)
        cp3[nt] = __builtin_amdgcn_wmma_f32_16x16x32_f16(false, a, false, b3f[nt],
                                                         (short)0, cp3[nt], false, false);
#pragma unroll
      for (int nt = 0; nt < 2; ++nt) cp3[nt] = fire_leak(cp3[nt], st, lane, nt);
    }

    // ---- layer 4: K=32, N=16 ----
    {
      v16h a = load_a_frag(st, lane, 0);
      cp4 = __builtin_amdgcn_wmma_f32_16x16x32_f16(false, a, false, b4f,
                                                   (short)0, cp4, false, false);
      cp4 = fire_leak(cp4, st, lane, 0);
    }

    // ---- layer 5: K=16(pad32, zero W cols), N=6(pad16, zero W rows) ----
    {
      v16h a = load_a_frag(st, lane, 0);   // cols 16..31 stale -> x0 weights
      v8f  d = __builtin_amdgcn_wmma_f32_16x16x32_f16(false, a, false, b5f,
                                                      (short)0, cp5, false, false);
      const int n  = lane & 15;
      const int mb = (lane & 16) ? 8 : 0;
      v8f cn;
#pragma unroll
      for (int r = 0; r < 8; ++r) {
        const float m    = d[r];
        const bool  fire = m > THRESH;
        const float s    = fire ? 1.0f : 0.0f;
        if (n < 6) ow[(mb + r)*6 + n] = s;
        cn[r] = BETA * m - s;
      }
      cp5 = cn;
    }

    // ---- coalesced per-wave output store: 16 samples x 6 = 96 floats ----
    {
      const size_t obase = ((size_t)t * Btot + b0w) * 6;
#pragma unroll
      for (int i = 0; i < 3; ++i)
        outg[obase + i * 32 + lane] = ow[i * 32 + lane];
    }
  }
}

extern "C" void kernel_launch(void* const* d_in, const int* in_sizes, int n_in,
                              void* d_out, int out_size, void* d_ws, size_t ws_size,
                              hipStream_t stream) {
  const float* x  = (const float*)d_in[0];
  const float* W1 = (const float*)d_in[1];
  const float* W2 = (const float*)d_in[2];
  const float* W3 = (const float*)d_in[3];
  const float* W4 = (const float*)d_in[4];
  const float* W5 = (const float*)d_in[5];
  float* out = (float*)d_out;

  const int Btot = in_sizes[0] / (TSTEPS * 10);   // x is (T, B, 10)
  int grid = Btot / 128;                          // 128 samples per workgroup
  if (grid < 1) grid = 1;

  snn_lif_wmma<<<grid, 256, 0, stream>>>(x, W1, W2, W3, W4, W5, out, Btot);
}
